// Attention2d_79173427135092
// MI455X (gfx1250) — matrix operand order repaired and checked by reading, not compiled, not run on hardware
//
#include <hip/hip_runtime.h>
#include <hip/hip_bf16.h>
#include <cstddef>
#include <cstdint>

// ---------------------------------------------------------------------------
// Attention2d for MI455X (gfx1250, wave32, WMMA).
//
// Roofline: ~17 GFLOP of matmul vs ~80MB of mandatory HBM traffic (rel_pos
// 64MB dominates; it fits in the 192MB L2 so the 4 batches hit L2). At
// 23.3 TB/s the kernel is memory-bound => bf16 WMMA (f32 accum) everywhere,
// flash-attention fusion so the 256MB score tensor never touches HBM.
//
// Round-1 changes:
//  * softmax row reductions now use DPP16 ROW_XMASK (VALU-only) instead of
//    __shfl_xor, which lowered to ds_bpermute_b32 and hammered the DS port
//    that also feeds WMMA operand loads.
//  * V chunk staging uses GLOBAL_LOAD_ASYNC_TO_LDS_B128 (per-lane 16B DMA,
//    ASYNCcnt) + s_wait_asynccnt, the CDNA5 async global->LDS path.
//
// WMMA layouts used (per CDNA5 ISA, wave32):
//   A 16x32 bf16 : lanes 0-15 -> M=lane, K in {0..7, 16..23} (8 VGPRs)
//                  lanes16-31 -> M=lane-16, K in {8..15, 24..31}
//   B 32x16 bf16 : lanes 0-15 -> N=lane, K in {0..7, 16..23} ; lanes16-31 -> N=lane-16,
//                  K in {8..15, 24..31} (same k order as A)
//   C/D 16x16 f32: lanes 0-15 -> N=lane, M=vgpr ; lanes16-31 -> M=vgpr+8
// ---------------------------------------------------------------------------

typedef __attribute__((ext_vector_type(16))) __bf16 v16bf;
typedef __attribute__((ext_vector_type(8)))  float  v8f;

struct U32x8 { uint4 a, b; };

static __device__ __forceinline__ v16bf mk16(uint4 a, uint4 b) {
    U32x8 t{a, b};
    return __builtin_bit_cast(v16bf, t);
}

// 16 contiguous bf16 from LDS -> v16bf (two ds_load_b128)
static __device__ __forceinline__ v16bf load16(const __bf16* p) {
    return mk16(*(const uint4*)p, *(const uint4*)(p + 8));
}

static __device__ __forceinline__ unsigned short f2bf_bits(float f) {
    unsigned u = __builtin_bit_cast(unsigned, f);
    u += 0x7FFFu + ((u >> 16) & 1u);          // round-to-nearest-even
    return (unsigned short)(u >> 16);
}
static __device__ __forceinline__ __bf16 f2bf(float f) {
    unsigned short b = f2bf_bits(f);
    return __builtin_bit_cast(__bf16, b);
}
static __device__ __forceinline__ __bf16 to_bf(float x)  { return f2bf(x); }
static __device__ __forceinline__ __bf16 to_bf(__bf16 x) { return x; }

static __device__ __forceinline__ void store_out(float* p, float v)  { *p = v; }
static __device__ __forceinline__ void store_out(__bf16* p, float v) { *p = f2bf(v); }

static __device__ __forceinline__ v8f wmma_bf16(v16bf a, v16bf b, v8f c) {
    // D = A(16x32 bf16) * B(32x16 bf16) + C(16x16 f32)
    return __builtin_amdgcn_wmma_f32_16x16x32_bf16(
        false, a, false, b, (short)0, c, false, false);
}

static __device__ __forceinline__ uint4 pack8_bf(v8f v) {
    uint4 o;
    o.x = (unsigned)f2bf_bits(v[0]) | ((unsigned)f2bf_bits(v[1]) << 16);
    o.y = (unsigned)f2bf_bits(v[2]) | ((unsigned)f2bf_bits(v[3]) << 16);
    o.z = (unsigned)f2bf_bits(v[4]) | ((unsigned)f2bf_bits(v[5]) << 16);
    o.w = (unsigned)f2bf_bits(v[6]) | ((unsigned)f2bf_bits(v[7]) << 16);
    return o;
}

// ---- DPP16 ROW_XMASK lane-xor within the 16-lane row (VALU-only shuffle) ---
template <int CTRL>
static __device__ __forceinline__ float dpp_f32(float x) {
    int r = __builtin_amdgcn_update_dpp(0, __builtin_bit_cast(int, x),
                                        CTRL, 0xF, 0xF, true);
    return __builtin_bit_cast(float, r);
}
static __device__ __forceinline__ float row_max16(float x) {
    x = fmaxf(x, dpp_f32<0x161>(x));   // row_xmask:1
    x = fmaxf(x, dpp_f32<0x162>(x));   // row_xmask:2
    x = fmaxf(x, dpp_f32<0x164>(x));   // row_xmask:4
    x = fmaxf(x, dpp_f32<0x168>(x));   // row_xmask:8
    return x;
}
static __device__ __forceinline__ float row_sum16(float x) {
    x += dpp_f32<0x161>(x);
    x += dpp_f32<0x162>(x);
    x += dpp_f32<0x164>(x);
    x += dpp_f32<0x168>(x);
    return x;
}

// ---- CDNA5 async global->LDS DMA (ASYNCcnt-tracked), 16B per lane ---------
static __device__ __forceinline__ void async_copy_b128(unsigned lds_byte_off,
                                                       const void* gaddr) {
    unsigned long long ga = (unsigned long long)gaddr;
    asm volatile("global_load_async_to_lds_b128 %0, %1, off"
                 :: "v"(lds_byte_off), "v"(ga) : "memory");
}
static __device__ __forceinline__ void wait_asynccnt0() {
    asm volatile("s_wait_asynccnt 0x0" ::: "memory");
}

// ---------------------------------------------------------------------------
// GEMM + bias:  Out[b, m, n] = sum_k W[m,k] * X[b,k,n] + bias[m]
// Tile: 64(M) x 128(N) per 256-thread WG (8 waves; wave w owns N-strip w*16,
// computes 4 stacked 16x16 D tiles). K stepped by 32 => 1 WMMA per tile/step.
// LDS staging: sW row-major [64][32] (pad 40), sX N-major [128][32] (pad 40)
// so every lane's A/B fragment is 2x b128 from LDS and global reads coalesce.
// ---------------------------------------------------------------------------
template <typename TIN, typename TOUT>
__global__ void __launch_bounds__(256)
gemm_bias_kernel(const float* __restrict__ W, const TIN* __restrict__ X,
                 const float* __restrict__ bias, TOUT* __restrict__ Out,
                 int Mtot, int Ktot, int Ntot) {
    const int b    = blockIdx.z;
    const int m0   = blockIdx.y * 64;
    const int n0   = blockIdx.x * 128;
    const int tid  = threadIdx.x;
    const int wave = tid >> 5;
    const int lane = tid & 31;
    const int half = lane >> 4;
    const int l16  = lane & 15;

    __shared__ __bf16 sW[64 * 40];    //  5.1 KB
    __shared__ __bf16 sX[128 * 40];   // 10.2 KB

    const TIN* Xb = X + (size_t)b * Ktot * Ntot;

    v8f acc[4];
#pragma unroll
    for (int mt = 0; mt < 4; ++mt) acc[mt] = {};

    for (int kk = 0; kk < Ktot; kk += 32) {
        __syncthreads();
        // W block [64 x 32] -> sW (coalesced: consecutive tid = consecutive k)
        for (int e = tid; e < 64 * 32; e += 256) {
            int m = e >> 5, c = e & 31;
            sW[m * 40 + c] = f2bf(W[(size_t)(m0 + m) * Ktot + kk + c]);
        }
        // X block [32 x 128] -> sX transposed (coalesced over n)
        for (int e = tid; e < 128 * 32; e += 256) {
            int n = e & 127, c = e >> 7;
            sX[n * 40 + c] = to_bf(Xb[(size_t)(kk + c) * Ntot + n0 + n]);
        }
        __syncthreads();

        // B fragment: lane -> column n0+wave*16+l16, K in {0..7,16..23} (+8 if half)
        v16bf Bt = mk16(*(const uint4*)(sX + (wave * 16 + l16) * 40 + half * 8), *(const uint4*)(sX + (wave * 16 + l16) * 40 + 16 + half * 8));
#pragma unroll
        for (int mt = 0; mt < 4; ++mt) {
            // A fragment: lane -> row mt*16+l16, K in {0..7,16..23} (+8 if half)
            const __bf16* ar = sW + (mt * 16 + l16) * 40 + half * 8;
            v16bf At = mk16(*(const uint4*)ar, *(const uint4*)(ar + 16));
            acc[mt] = wmma_bf16(At, Bt, acc[mt]);
        }
    }

    // Epilogue: bias + store. Lanes of one half-wave cover 16 consecutive n
    // => 64B (f32) / 32B (bf16) coalesced stores per row.
#pragma unroll
    for (int mt = 0; mt < 4; ++mt) {
#pragma unroll
        for (int r = 0; r < 8; ++r) {
            int M = m0 + mt * 16 + half * 8 + r;
            int n = n0 + wave * 16 + l16;
            float v = acc[mt][r] + bias[M];
            store_out(Out + ((size_t)b * Mtot + M) * Ntot + n, v);
        }
    }
}

// ---------------------------------------------------------------------------
// Fused flash attention per (b, h): S = Qt*K*scale + rel, online softmax,
// O = P * Vt. One WG (8 waves) per 128 query rows; each wave owns 16 rows.
// K/V streamed through LDS in 256-column chunks (static LDS ~57 KB).
// qkv scratch layout: [B][1536][1024] bf16, head h rows h*96 + {q:0,k:32,v:64}.
// ---------------------------------------------------------------------------
__global__ void __launch_bounds__(256)
attn_kernel(const __bf16* __restrict__ qkv, const float* __restrict__ rel,
            __bf16* __restrict__ aout) {
    const int b    = blockIdx.z;
    const int h    = blockIdx.y;
    const int i0   = blockIdx.x * 128;
    const int tid  = threadIdx.x;
    const int wave = tid >> 5;
    const int lane = tid & 31;
    const int half = lane >> 4;
    const int l16  = lane & 15;

    __shared__ __bf16 sKt[256 * 40];      // K chunk, column-major-in-d, 20.5 KB
    __shared__ __bf16 sVs[32 * 264];      // V chunk, row-major per d,   16.9 KB
    __shared__ __bf16 sQt[128 * 40];      // Q block transposed,         10.2 KB
    __shared__ __bf16 sPs[8 * 16 * 40];   // per-wave P relayout tile,   10.2 KB

    const __bf16* qp = qkv + (((size_t)b * 1536) + h * 96) * 1024;
    const __bf16* kp = qp + 32 * 1024;
    const __bf16* vp = qp + 64 * 1024;

    // Stage Q block transposed: sQt[i][d]
    for (int e = tid; e < 32 * 128; e += 256) {
        int d = e >> 7, i = e & 127;
        sQt[i * 40 + d] = qp[d * 1024 + i0 + i];
    }
    __syncthreads();
    // A fragment of Qt (16 rows x K=32 = dh) — loaded once, reused 64 times.
    const __bf16* qrow = sQt + (wave * 16 + l16) * 40 + half * 8;
    const v16bf Qa = mk16(*(const uint4*)qrow, *(const uint4*)(qrow + 16));

    v8f O0 = {}, O1 = {};
    float mrow[8], lrow[8];
#pragma unroll
    for (int r = 0; r < 8; ++r) { mrow[r] = -1e30f; lrow[r] = 0.0f; }

    const float scale = 0.17677669529663687f;   // 32^-0.5
    const float* relbase =
        rel + ((size_t)h * 1024 + (i0 + wave * 16 + half * 8)) * 1024 + l16;

    for (int jc = 0; jc < 1024; jc += 256) {
        __syncthreads();
        // V chunk: rows stay row-major => pure byte copy. Use the CDNA5
        // async DMA path: 16B per lane-op, tracked by ASYNCcnt (no VGPR
        // round-trip, overlaps with the K transpose staging below).
        for (int c = tid; c < 1024; c += 256) {          // 32 rows x 32 chunks
            int d = c >> 5, nc = c & 31;
            async_copy_b128((unsigned)(uintptr_t)(sVs + d * 264 + nc * 8),
                            (const void*)(vp + (size_t)d * 1024 + jc + nc * 8));
        }
        // K chunk: transposed staging sKt[n][d] (element scatter, b16 stores)
        for (int e = tid; e < 32 * 256; e += 256) {
            int d = e >> 8, n = e & 255;
            sKt[n * 40 + d] = kp[d * 1024 + jc + n];
        }
        wait_asynccnt0();          // V DMA complete (per-wave) ...
        __syncthreads();           // ... then make all staging visible to WG

        for (int j2 = 0; j2 < 256; j2 += 32) {
            // S tiles: one WMMA each (K == dh == 32)
            v16bf Kb0 = mk16(*(const uint4*)(sKt + (j2 + l16) * 40 + half * 8), *(const uint4*)(sKt + (j2 + l16) * 40 + 16 + half * 8));
            v16bf Kb1 = mk16(*(const uint4*)(sKt + (j2 + 16 + l16) * 40 + half * 8), *(const uint4*)(sKt + (j2 + 16 + l16) * 40 + 16 + half * 8));
            v8f z = {};
            v8f S0 = wmma_bf16(Qa, Kb0, z);
            v8f S1 = wmma_bf16(Qa, Kb1, z);

            // scale + rel_pos (the single real HBM stream: 64B coalesced rows)
            const float* rp = relbase + jc + j2;
#pragma unroll
            for (int r = 0; r < 8; ++r) {
                S0[r] = S0[r] * scale + rp[(size_t)r * 1024];
                S1[r] = S1[r] * scale + rp[(size_t)r * 1024 + 16];
            }

            // Online softmax. D layout: lane = column, VGPR r = row half*8+r,
            // so row reduction = DPP ROW_XMASK over the 16-lane half
            // (VALU-only; keeps the DS port free for WMMA fragment loads).
#pragma unroll
            for (int r = 0; r < 8; ++r) {
                float t  = row_max16(fmaxf(S0[r], S1[r]));
                float mn = fmaxf(mrow[r], t);
                float al = __expf(mrow[r] - mn);
                mrow[r] = mn;
                S0[r] = __expf(S0[r] - mn);
                S1[r] = __expf(S1[r] - mn);
                float rs = row_sum16(S0[r] + S1[r]);
                lrow[r] = lrow[r] * al + rs;
                O0[r] *= al;
                O1[r] *= al;
            }

            // D-layout f32 P  ->  A-layout bf16 P via per-wave LDS tile.
            // Same-wave LDS RAW ordered by DScnt (compiler inserts waits).
            __bf16* pw = sPs + wave * (16 * 40);
#pragma unroll
            for (int r = 0; r < 8; ++r) {
                int row = half * 8 + r;
                pw[row * 40 + l16]      = f2bf(S0[r]);
                pw[row * 40 + 16 + l16] = f2bf(S1[r]);
            }
            const __bf16* prow = pw + l16 * 40 + half * 8;
            v16bf Pa = mk16(*(const uint4*)prow, *(const uint4*)(prow + 16));

            // O += P(16x32) * Vt(32x32) : B fragments from sVs rows (contig k)
            v16bf Vb0 = mk16(*(const uint4*)(sVs + l16 * 264 + j2 + half * 8), *(const uint4*)(sVs + l16 * 264 + j2 + 16 + half * 8));
            v16bf Vb1 = mk16(*(const uint4*)(sVs + (16 + l16) * 264 + j2 + half * 8), *(const uint4*)(sVs + (16 + l16) * 264 + j2 + 16 + half * 8));
            O0 = wmma_bf16(Pa, Vb0, O0);
            O1 = wmma_bf16(Pa, Vb1, O1);
        }
    }

#pragma unroll
    for (int r = 0; r < 8; ++r) {
        float inv = 1.0f / lrow[r];
        O0[r] *= inv;
        O1[r] *= inv;
    }

    // Store: lane owns column d = l16 (+16), 8 consecutive query rows
    // => pack to one 16B store per tile. Channel = h*32 + d.
    size_t nbase = (size_t)i0 + wave * 16 + half * 8;
    __bf16* o0p = aout + (((size_t)b * 512) + h * 32 + l16) * 1024 + nbase;
    __bf16* o1p = o0p + (size_t)16 * 1024;
    *(uint4*)o0p = pack8_bf(O0);
    *(uint4*)o1p = pack8_bf(O1);
}

// ---------------------------------------------------------------------------
// Host launcher. Inputs (all f32): x, w_qkv, b_qkv, w_proj, b_proj, rel_pos.
// Workspace: qkv bf16 [4][1536][1024] (12.6MB) + attnout bf16 [4][512][1024]
// (4.2MB) = ~16.8MB.
// ---------------------------------------------------------------------------
extern "C" void kernel_launch(void* const* d_in, const int* in_sizes, int n_in,
                              void* d_out, int out_size, void* d_ws, size_t ws_size,
                              hipStream_t stream) {
    (void)in_sizes; (void)n_in; (void)out_size; (void)ws_size;
    const float* x      = (const float*)d_in[0];
    const float* w_qkv  = (const float*)d_in[1];
    const float* b_qkv  = (const float*)d_in[2];
    const float* w_proj = (const float*)d_in[3];
    const float* b_proj = (const float*)d_in[4];
    const float* rel    = (const float*)d_in[5];
    float*       out    = (float*)d_out;

    __bf16* qkv     = (__bf16*)d_ws;
    __bf16* attnout = qkv + (size_t)4 * 1536 * 1024;

    dim3 blk(256);

    // 1) QKV projection: [1536x512] x [512x1024] per batch
    gemm_bias_kernel<float, __bf16>
        <<<dim3(1024 / 128, 1536 / 64, 4), blk, 0, stream>>>(
            w_qkv, x, b_qkv, qkv, 1536, 512, 1024);

    // 2) Fused flash attention: (b, h, 128-row query block)
    attn_kernel<<<dim3(1024 / 128, 16, 4), blk, 0, stream>>>(qkv, rel, attnout);

    // 3) Output projection: [512x512] x [512x1024] per batch -> f32 out
    gemm_bias_kernel<__bf16, float>
        <<<dim3(1024 / 128, 512 / 64, 4), blk, 0, stream>>>(
            w_proj, attnout, b_proj, out, 512, 512, 1024);
}
